// MLA_CrossAttention_41326175322767
// MI455X (gfx1250) — compile-verified, matrix operand order checked by reading, not run on hardware
//
#include <hip/hip_runtime.h>
#include <math.h>

// ---------------------------------------------------------------------------
// MLA cross-attention for MI455X (gfx1250), bf16 WMMA compute / fp32 accumulate
// All GEMMs are edge-free: weights are pre-transposed/padded/bf16-converted,
// activations padded so every tile load is unconditional and vectorized.
// ---------------------------------------------------------------------------

#define B_   2
#define T_   2048
#define S_   2048
#define D_   1024
#define H_   16
#define HD_  64
#define RHD_ 32
#define NHD_ 32
#define QLAT_  256
#define KVLAT_ 204
#define KVPAD_ 256   // KV_LAT padded to K-chunk multiple

typedef __attribute__((ext_vector_type(16))) __bf16 v16bf;
typedef __attribute__((ext_vector_type(8)))  __bf16 v8bf;
typedef __attribute__((ext_vector_type(4)))  __bf16 v4bf;
typedef __attribute__((ext_vector_type(8)))  float  v8f;
typedef __attribute__((ext_vector_type(4)))  float  v4f;

static __device__ inline v8f vzero8f() {
  v8f z = {0.f,0.f,0.f,0.f,0.f,0.f,0.f,0.f};
  return z;
}

// Load one 16-bit WMMA fragment: 8 contiguous bf16 at p0 (k = kbase..kbase+7)
// and 8 contiguous bf16 at p1 (k = kbase+16..kbase+23), per ISA 7.12.2.
static __device__ inline v16bf ld_frag(const __bf16* p0, const __bf16* p1) {
  v8bf lo = *(const v8bf*)p0;
  v8bf hi = *(const v8bf*)p1;
  return __builtin_shufflevector(lo, hi, 0,1,2,3,4,5,6,7,8,9,10,11,12,13,14,15);
}

static __device__ inline v8f wmma_bf16(v16bf a, v16bf b, v8f c) {
  // (neg_a, A, neg_b, B, c_mod, C, reuse_a, reuse_b)
  return __builtin_amdgcn_wmma_f32_16x16x32_bf16(false, a, false, b, (short)0, c,
                                                 false, false);
}

static __device__ inline float redmax16(float v) {
  v = fmaxf(v, __shfl_xor(v, 1, 32));
  v = fmaxf(v, __shfl_xor(v, 2, 32));
  v = fmaxf(v, __shfl_xor(v, 4, 32));
  v = fmaxf(v, __shfl_xor(v, 8, 32));
  return v;
}
static __device__ inline float redsum16(float v) {
  v += __shfl_xor(v, 1, 32);
  v += __shfl_xor(v, 2, 32);
  v += __shfl_xor(v, 4, 32);
  v += __shfl_xor(v, 8, 32);
  return v;
}

// ---------------------------------------------------------------------------
// Weight reformat (one-off, tiny): Wt[n][k] (bf16, Np x Kp, zero-padded) from
// W[k][n] (f32, K x N).
// ---------------------------------------------------------------------------
__global__ __launch_bounds__(256)
void wtranspose_kernel(const float* __restrict__ W, __bf16* __restrict__ Wt,
                       int K, int N, int Kp) {
  int idx = blockIdx.x * 256 + threadIdx.x;   // over Np*Kp, k fastest
  int n = idx / Kp;
  int k = idx % Kp;
  float v = (k < K && n < N) ? W[(size_t)k * N + n] : 0.0f;
  Wt[idx] = (__bf16)v;
}

// ---------------------------------------------------------------------------
// GEMM: C[M, Np] = A[M, K] (f32) x Wt[Np, K]^T (bf16). M % 64 == 0,
// Np % 128 == 0, K % 64 == 0 — fully edge-free.
// Block tile 64(M) x 128(N), 8 waves, K-chunks of 64 staged through LDS.
// ---------------------------------------------------------------------------
__global__ __launch_bounds__(256)
void gemm_bf16_kernel(const float* __restrict__ A, const __bf16* __restrict__ Wt,
                      const float* __restrict__ bias, float* __restrict__ C,
                      int K, int ldA, int ldWt, int ldC) {
  __shared__ __bf16 sA[64 * 64];    // [m][k]
  __shared__ __bf16 sW[128 * 64];   // [n][k]

  const int tid   = threadIdx.x;
  const int wave  = tid >> 5;
  const int lane  = tid & 31;
  const int m0    = blockIdx.y * 64;
  const int n0    = blockIdx.x * 128;
  const int mSub  = wave & 3;   // 16-row slice
  const int nHalf = wave >> 2;  // 64-col half
  const int mL    = lane & 15;
  const int kb    = (lane < 16) ? 0 : 8;

  v8f acc[4];
  #pragma unroll
  for (int j = 0; j < 4; ++j) acc[j] = vzero8f();

  for (int kc = 0; kc < K; kc += 64) {
    // Stage A 64x64 f32 -> bf16: 4 x float4 per thread, unconditional.
    #pragma unroll
    for (int i = 0; i < 4; ++i) {
      int v  = tid + i * 256;          // 0..1023 vec4 slots
      int r  = v >> 4;                 // 16 vec4 per row
      int c4 = (v & 15) << 2;
      v4f d = *(const v4f*)(A + (size_t)(m0 + r) * ldA + kc + c4);
      v4bf b;
      b[0] = (__bf16)d[0]; b[1] = (__bf16)d[1];
      b[2] = (__bf16)d[2]; b[3] = (__bf16)d[3];
      *(v4bf*)(sA + r * 64 + c4) = b;
    }
    // Stage W 128x64 bf16: straight 16B copies from pre-transposed weights.
    #pragma unroll
    for (int i = 0; i < 4; ++i) {
      int v  = tid + i * 256;          // 0..1023 v8 slots
      int n  = v >> 3;                 // 8 v8 per row
      int k8 = (v & 7) << 3;
      *(v8bf*)(sW + n * 64 + k8) =
          *(const v8bf*)(Wt + (size_t)(n0 + n) * ldWt + kc + k8);
    }
    __syncthreads();

    #pragma unroll
    for (int c = 0; c < 2; ++c) {
      const __bf16* ap = sA + (mSub * 16 + mL) * 64 + c * 32 + kb;
      v16bf af = ld_frag(ap, ap + 16);
      #pragma unroll
      for (int j = 0; j < 4; ++j) {
        int nl = nHalf * 64 + j * 16 + mL;
        const __bf16* bp = sW + nl * 64 + c * 32 + kb;
        acc[j] = wmma_bf16(af, ld_frag(bp, bp + 16), acc[j]);
      }
    }
    __syncthreads();
  }

  #pragma unroll
  for (int j = 0; j < 4; ++j) {
    int n = n0 + nHalf * 64 + j * 16 + mL;
    float bv = bias ? bias[n] : 0.0f;
    #pragma unroll
    for (int r = 0; r < 8; ++r) {
      int m = m0 + mSub * 16 + r + ((lane < 16) ? 0 : 8);
      C[(size_t)m * ldC + n] = acc[j][r] + bv;
    }
  }
}

// ---------------------------------------------------------------------------
// LayerNorm (in place), one block per row; ld may exceed Cc (padded buffers).
// ---------------------------------------------------------------------------
__global__ __launch_bounds__(256)
void ln_kernel(float* __restrict__ X, const float* __restrict__ g,
               const float* __restrict__ b, int Cc, int ld) {
  __shared__ float r1[256];
  __shared__ float r2[256];
  const int row = blockIdx.x;
  const int tid = threadIdx.x;
  float* x = X + (size_t)row * ld;

  float s = 0.f, s2 = 0.f;
  for (int c = tid; c < Cc; c += 256) { float v = x[c]; s += v; s2 += v * v; }
  r1[tid] = s; r2[tid] = s2;
  __syncthreads();
  for (int off = 128; off > 0; off >>= 1) {
    if (tid < off) { r1[tid] += r1[tid + off]; r2[tid] += r2[tid + off]; }
    __syncthreads();
  }
  float mean = r1[0] / (float)Cc;
  float var  = r2[0] / (float)Cc - mean * mean;
  float inv  = rsqrtf(var + 1e-5f);
  for (int c = tid; c < Cc; c += 256)
    x[c] = (x[c] - mean) * inv * g[c] + b[c];
}

// ---------------------------------------------------------------------------
// RoPE helpers: freq(i) = 10000^(-i/16) = exp2(-i * log2(10000)/16)
// ---------------------------------------------------------------------------
static __device__ inline void rope_cs(int pos, int i, float* cs, float* sn) {
  float freq = exp2f(-(float)i * 0.8304820237218406f);
  float ang = (float)pos * freq;
  __sincosf(ang, sn, cs);
}

// Assemble Q (B,H,T,64) bf16 from q_nope (B,T,H*32) and rope(q_rope).
__global__ __launch_bounds__(256)
void assemble_q_kernel(const float* __restrict__ qn, const float* __restrict__ qr,
                       __bf16* __restrict__ Q) {
  int idx = blockIdx.x * 256 + threadIdx.x;     // B*T*H*64 total
  int d = idx & 63;
  int h = (idx >> 6) & (H_ - 1);
  int t = (idx >> 10) & (T_ - 1);
  int b = idx >> 21;
  float val;
  if (d < NHD_) {
    val = qn[((size_t)(b * T_ + t)) * (H_ * NHD_) + h * NHD_ + d];
  } else {
    int dd = d - NHD_;
    int i = dd >> 1;
    size_t base = ((size_t)(b * T_ + t)) * (H_ * RHD_) + h * RHD_ + 2 * i;
    float ev = qr[base], od = qr[base + 1];
    float cs, sn; rope_cs(t, i, &cs, &sn);
    val = (dd & 1) ? (ev * sn + od * cs) : (ev * cs - od * sn);
  }
  Q[((size_t)(b * H_ + h) * T_ + t) * HD_ + d] = (__bf16)val;
}

// Assemble K (B,H,S,64) bf16 from k_nope (B,S,H*32) and rope(k_rope)/H.
// kr has row stride 128 (padded N).
__global__ __launch_bounds__(256)
void assemble_k_kernel(const float* __restrict__ kn, const float* __restrict__ kr,
                       __bf16* __restrict__ Kd) {
  int idx = blockIdx.x * 256 + threadIdx.x;
  int d = idx & 63;
  int h = (idx >> 6) & (H_ - 1);
  int s = (idx >> 10) & (S_ - 1);
  int b = idx >> 21;
  float val;
  if (d < NHD_) {
    val = kn[((size_t)(b * S_ + s)) * (H_ * NHD_) + h * NHD_ + d];
  } else {
    int dd = d - NHD_;
    int i = dd >> 1;
    size_t base = ((size_t)(b * S_ + s)) * 128 + 2 * i;
    float ev = kr[base], od = kr[base + 1];
    float cs, sn; rope_cs(s, i, &cs, &sn);
    val = ((dd & 1) ? (ev * sn + od * cs) : (ev * cs - od * sn)) * (1.0f / H_);
  }
  Kd[((size_t)(b * H_ + h) * S_ + s) * HD_ + d] = (__bf16)val;
}

// Tiled transpose: Vt[b][h][d][s] (bf16) from vtmp (B,S,H*64) f32.
__global__ __launch_bounds__(256)
void transpose_v_kernel(const float* __restrict__ vtmp, __bf16* __restrict__ Vt) {
  __shared__ __bf16 tile[32][33];
  const int bh = blockIdx.x;          // B*H
  const int s0 = blockIdx.y * 32;
  const int d0 = blockIdx.z * 32;
  const int b = bh >> 4, h = bh & 15;
  #pragma unroll
  for (int i = 0; i < 4; ++i) {
    int s = s0 + threadIdx.y + i * 8;
    int d = d0 + threadIdx.x;
    tile[threadIdx.y + i * 8][threadIdx.x] =
        (__bf16)vtmp[((size_t)(b * S_ + s)) * (H_ * HD_) + h * HD_ + d];
  }
  __syncthreads();
  #pragma unroll
  for (int i = 0; i < 4; ++i) {
    int d = d0 + threadIdx.y + i * 8;
    int s = s0 + threadIdx.x;
    Vt[((size_t)bh * HD_ + d) * S_ + s] = tile[threadIdx.x][threadIdx.y + i * 8];
  }
}

// ---------------------------------------------------------------------------
// Flash attention: grid (B*H, T/128), 128 threads (4 waves).
// Each wave owns 32 query rows (two 16-row tiles) so every K / V^T B-fragment
// feeds two WMMAs. Online softmax in exp2 domain (v_exp_f32 is natively exp2).
// P staged through per-wave LDS (per-wave DS ordering via s_wait_dscnt).
// ---------------------------------------------------------------------------
__global__ __launch_bounds__(128)
void attn_kernel(const __bf16* __restrict__ Q, const __bf16* __restrict__ Kd,
                 const __bf16* __restrict__ Vt, float* __restrict__ O) {
  __shared__ __bf16 sP[4][32 * 32];

  const int bh   = blockIdx.x;
  const int b    = bh >> 4, h = bh & 15;
  const int wave = threadIdx.x >> 5;
  const int lane = threadIdx.x & 31;
  const int t0   = blockIdx.y * 128 + wave * 32;
  const int mL   = lane & 15;
  const int kb   = (lane < 16) ? 0 : 8;
  const int rOff = (lane < 16) ? 0 : 8;

  const __bf16* q  = Q  + (size_t)bh * T_ * HD_;
  const __bf16* k  = Kd + (size_t)bh * S_ * HD_;
  const __bf16* vt = Vt + (size_t)bh * HD_ * S_;
  __bf16* myP = sP[wave];

  // Q fragments: two 16-row tiles x two 32-wide k-chunks, resident in VGPRs.
  v16bf qf[2][2];
  #pragma unroll
  for (int u = 0; u < 2; ++u)
    #pragma unroll
    for (int c = 0; c < 2; ++c) {
      const __bf16* base = q + (size_t)(t0 + u * 16 + mL) * HD_ + c * 32 + kb;
      qf[u][c] = ld_frag(base, base + 16);
    }

  float mrow[2][8], lrow[2][8];
  #pragma unroll
  for (int u = 0; u < 2; ++u)
    #pragma unroll
    for (int r = 0; r < 8; ++r) { mrow[u][r] = -1e30f; lrow[u][r] = 0.f; }
  v8f acc[2][4];
  #pragma unroll
  for (int u = 0; u < 2; ++u)
    #pragma unroll
    for (int j = 0; j < 4; ++j) acc[u][j] = vzero8f();

  // 1/sqrt(64) * log2(e): softmax runs in the exp2 domain.
  const float sc = 0.125f * 1.4426950408889634f;

  for (int s0 = 0; s0 < S_; s0 += 32) {
    if (s0 + 32 < S_) {
      __builtin_prefetch(k + (size_t)(s0 + 32 + mL) * HD_, 0, 1);
      __builtin_prefetch(vt + (size_t)mL * S_ + s0 + 32, 0, 1);
    }
    // K fragments for this 32-key chunk, shared by both row tiles.
    v16bf kf0[2], kf1[2];
    #pragma unroll
    for (int c = 0; c < 2; ++c) {
      const __bf16* p0 = k + (size_t)(s0 + mL) * HD_ + c * 32 + kb;
      kf0[c] = ld_frag(p0, p0 + 16);
      const __bf16* p1 = k + (size_t)(s0 + 16 + mL) * HD_ + c * 32 + kb;
      kf1[c] = ld_frag(p1, p1 + 16);
    }

    #pragma unroll
    for (int u = 0; u < 2; ++u) {
      v8f c0 = vzero8f(), c1 = vzero8f();
      #pragma unroll
      for (int c = 0; c < 2; ++c) {
        c0 = wmma_bf16(qf[u][c], kf0[c], c0);
        c1 = wmma_bf16(qf[u][c], kf1[c], c1);
      }
      // Online softmax update (log2 domain).
      #pragma unroll
      for (int r = 0; r < 8; ++r) {
        float s0v = c0[r] * sc, s1v = c1[r] * sc;
        float nm = fmaxf(mrow[u][r], redmax16(fmaxf(s0v, s1v)));
        float p0 = exp2f(s0v - nm);
        float p1 = exp2f(s1v - nm);
        float alpha = exp2f(mrow[u][r] - nm);
        lrow[u][r] = lrow[u][r] * alpha + redsum16(p0 + p1);
        mrow[u][r] = nm;
        #pragma unroll
        for (int j = 0; j < 4; ++j) acc[u][j][r] *= alpha;
        int m = u * 16 + r + rOff;
        myP[m * 32 + mL]      = (__bf16)p0;
        myP[m * 32 + 16 + mL] = (__bf16)p1;
      }
    }
    // Per-wave DS ordering: stores above complete before fragment loads below.
    asm volatile("s_wait_dscnt 0" ::: "memory");

    // P (2 x 16x32) x V^T slices; each V fragment feeds both row tiles.
    v16bf pf[2];
    #pragma unroll
    for (int u = 0; u < 2; ++u) {
      const __bf16* pbase = myP + (u * 16 + mL) * 32 + kb;
      pf[u] = ld_frag(pbase, pbase + 16);
    }
    #pragma unroll
    for (int j = 0; j < 4; ++j) {
      const __bf16* vb = vt + (size_t)(j * 16 + mL) * S_ + s0 + kb;
      v16bf vf = ld_frag(vb, vb + 16);
      acc[0][j] = wmma_bf16(pf[0], vf, acc[0][j]);
      acc[1][j] = wmma_bf16(pf[1], vf, acc[1][j]);
    }
  }

  // Epilogue: normalize and scatter to o (B,T,H*64) f32.
  #pragma unroll
  for (int u = 0; u < 2; ++u)
    #pragma unroll
    for (int j = 0; j < 4; ++j) {
      #pragma unroll
      for (int r = 0; r < 8; ++r) {
        int t = t0 + u * 16 + r + rOff;
        int d = j * 16 + mL;
        O[((size_t)(b * T_ + t)) * (H_ * HD_) + h * HD_ + d] =
            acc[u][j][r] / lrow[u][r];
      }
    }
}

// ---------------------------------------------------------------------------
// Host launcher
// ---------------------------------------------------------------------------
extern "C" void kernel_launch(void* const* d_in, const int* in_sizes, int n_in,
                              void* d_out, int out_size, void* d_ws, size_t ws_size,
                              hipStream_t stream) {
  (void)in_sizes; (void)n_in; (void)out_size; (void)ws_size;
  const float* x       = (const float*)d_in[0];
  const float* xf      = (const float*)d_in[1];
  const float* Wqc     = (const float*)d_in[2];
  const float* gq      = (const float*)d_in[3];
  const float* bq      = (const float*)d_in[4];
  const float* Wq_rope = (const float*)d_in[5];
  const float* Wq_nope = (const float*)d_in[6];
  const float* Wkvc    = (const float*)d_in[7];
  const float* gkv     = (const float*)d_in[8];
  const float* bkv     = (const float*)d_in[9];
  const float* Wk_nope = (const float*)d_in[10];
  const float* Wv      = (const float*)d_in[11];
  const float* Wkr     = (const float*)d_in[12];
  const float* Wo      = (const float*)d_in[13];
  const float* bo      = (const float*)d_in[14];
  float* out = (float*)d_out;

  const int BT = B_ * T_;   // 4096 rows
  size_t off = 0;
  auto alloc = [&](size_t bytes) {
    size_t o = off; off += (bytes + 255) & ~(size_t)255; return o;
  };
  char* ws = (char*)d_ws;
  // Activations (f32, padded leading dims).
  float*  ql   = (float*)(ws + alloc((size_t)BT * QLAT_ * 4));
  float*  kvl  = (float*)(ws + alloc((size_t)BT * KVPAD_ * 4));   // 204 -> 256
  float*  qn   = (float*)(ws + alloc((size_t)BT * H_ * NHD_ * 4));
  float*  qr   = (float*)(ws + alloc((size_t)BT * H_ * RHD_ * 4));
  float*  kn   = (float*)(ws + alloc((size_t)BT * H_ * NHD_ * 4));
  float*  kr   = (float*)(ws + alloc((size_t)BT * 128 * 4));      // 32 -> 128
  float*  vtmp = (float*)(ws + alloc((size_t)BT * H_ * HD_ * 4));
  float*  obuf = (float*)(ws + alloc((size_t)BT * H_ * HD_ * 4));
  // Attention operands (bf16).
  __bf16* Qb   = (__bf16*)(ws + alloc((size_t)B_ * H_ * T_ * HD_ * 2));
  __bf16* Kb   = (__bf16*)(ws + alloc((size_t)B_ * H_ * S_ * HD_ * 2));
  __bf16* Vtb  = (__bf16*)(ws + alloc((size_t)B_ * H_ * HD_ * S_ * 2));
  // Pre-transposed / padded bf16 weights: Wt[Np][Kp].
  __bf16* Wqc_t = (__bf16*)(ws + alloc((size_t)256  * 1024 * 2));
  __bf16* Wqn_t = (__bf16*)(ws + alloc((size_t)512  * 256  * 2));
  __bf16* Wqr_t = (__bf16*)(ws + alloc((size_t)512  * 256  * 2));
  __bf16* Wkvc_t= (__bf16*)(ws + alloc((size_t)256  * 1024 * 2));
  __bf16* Wkn_t = (__bf16*)(ws + alloc((size_t)512  * 256  * 2));
  __bf16* Wv_t  = (__bf16*)(ws + alloc((size_t)1024 * 256  * 2));
  __bf16* Wkr_t = (__bf16*)(ws + alloc((size_t)128  * 1024 * 2));
  __bf16* Wo_t  = (__bf16*)(ws + alloc((size_t)1024 * 1024 * 2));

  auto wtr = [&](const float* W, __bf16* Wt, int K, int N, int Kp, int Np) {
    wtranspose_kernel<<<(Np * Kp) / 256, 256, 0, stream>>>(W, Wt, K, N, Kp);
  };
  auto gemm = [&](const float* A, const __bf16* Wt, const float* bias, float* C,
                  int Np, int K, int ldA, int ldWt, int ldC) {
    dim3 grid(Np / 128, BT / 64);
    gemm_bf16_kernel<<<grid, 256, 0, stream>>>(A, Wt, bias, C, K, ldA, ldWt, ldC);
  };

  // 0) Reformat all weights once (transpose + pad + bf16).
  wtr(Wqc,     Wqc_t, 1024, 256,  1024, 256);
  wtr(Wq_nope, Wqn_t, 256,  512,  256,  512);
  wtr(Wq_rope, Wqr_t, 256,  512,  256,  512);
  wtr(Wkvc,    Wkvc_t,1024, 204,  1024, 256);
  wtr(Wk_nope, Wkn_t, 204,  512,  256,  512);
  wtr(Wv,      Wv_t,  204,  1024, 256,  1024);
  wtr(Wkr,     Wkr_t, 1024, 32,   1024, 128);
  wtr(Wo,      Wo_t,  1024, 1024, 1024, 1024);

  // 1) ql = LN(x @ Wqc)
  gemm(x, Wqc_t, nullptr, ql, 256, 1024, D_, 1024, QLAT_);
  ln_kernel<<<BT, 256, 0, stream>>>(ql, gq, bq, QLAT_, QLAT_);
  // 2) kvl = LN(xf @ Wkvc)   (pad cols 204..255 become exact zeros)
  gemm(xf, Wkvc_t, nullptr, kvl, 256, 1024, D_, 1024, KVPAD_);
  ln_kernel<<<BT, 256, 0, stream>>>(kvl, gkv, bkv, KVLAT_, KVPAD_);
  // 3) q_nope / q_rope projections (K = 256)
  gemm(ql, Wqn_t, nullptr, qn, 512, QLAT_, QLAT_, 256, H_ * NHD_);
  gemm(ql, Wqr_t, nullptr, qr, 512, QLAT_, QLAT_, 256, H_ * RHD_);
  // 4) k_nope / v / k_rope projections (K padded 204 -> 256)
  gemm(kvl, Wkn_t, nullptr, kn, 512, KVPAD_, KVPAD_, 256, H_ * NHD_);
  gemm(kvl, Wv_t, nullptr, vtmp, 1024, KVPAD_, KVPAD_, 256, H_ * HD_);
  gemm(xf, Wkr_t, nullptr, kr, 128, 1024, D_, 1024, 128);
  // 5) RoPE + layout to bf16 attention operands
  assemble_q_kernel<<<(B_ * T_ * H_ * HD_) / 256, 256, 0, stream>>>(qn, qr, Qb);
  assemble_k_kernel<<<(B_ * S_ * H_ * HD_) / 256, 256, 0, stream>>>(kn, kr, Kb);
  transpose_v_kernel<<<dim3(B_ * H_, S_ / 32, HD_ / 32), dim3(32, 8), 0, stream>>>(vtmp, Vtb);
  // 6) Flash attention (32 query rows per wave)
  attn_kernel<<<dim3(B_ * H_, T_ / 128), 128, 0, stream>>>(Qb, Kb, Vtb, obuf);
  // 7) out = o @ Wo + bo
  gemm(obuf, Wo_t, bo, out, 1024, 1024, 1024, 1024, 1024);
}
